// LegislativeGraphModel_61607010893928
// MI455X (gfx1250) — matrix operand order repaired and checked by reading, not compiled, not run on hardware
//
#include <hip/hip_runtime.h>
#include <hip/hip_bf16.h>

// ---------------- problem constants ----------------
#define NA 8192
#define NB 2048
#define DIM 192
#define NH 4
#define DK 48
#define DKP 64                  // per-head K padded to 64 for 2x wmma K=32
#define SEG 4                   // bill-dimension split for parallelism
#define BSEG (NB / SEG)         // 512 bills per segment
#define CHUNK 32                // bills per flash chunk
#define NCH (BSEG / CHUNK)      // 16 chunks per segment (even)

typedef __attribute__((ext_vector_type(16))) __bf16 v16bf;
typedef __attribute__((ext_vector_type(8)))  float  v8f;
typedef unsigned short u16;

template <bool B> struct BoolC { static constexpr bool value = B; };

// ---------------- helpers ----------------
__device__ __forceinline__ u16 f2bf(float f) {
  unsigned u = __float_as_uint(f);
  u = (u + 0x7FFFu + ((u >> 16) & 1u)) >> 16;   // RNE
  return (u16)u;
}

__device__ __forceinline__ float fix_io(float v) {  // nan_to_num(nan=0, posinf=1, neginf=-1)
  if (v != v) return 0.f;
  if (v == __builtin_inff()) return 1.f;
  if (v == -__builtin_inff()) return -1.f;
  return v;
}

// A operand: 16x32 bf16, M rows in memory (row-major, rowStride in halves).
// lane l: m = l&15, kb = (l>>4)*8; elems 0..7 = K kb..kb+7, elems 8..15 = K kb+16..kb+23
__device__ __forceinline__ v16bf loadA16(const u16* __restrict__ base, int rowStride, int lane) {
  int m = lane & 15, kb = (lane >> 4) << 3;
  union { v16bf v; uint4 q[2]; } u;
  const u16* r = base + (size_t)m * rowStride + kb;
  u.q[0] = *(const uint4*)(r);
  u.q[1] = *(const uint4*)(r + 16);
  return u.v;
}

// B operand: 32x16 bf16 (KxN); memory holds B^T row-major (row n = column n of B).
// lane l: n = l&15, kb = (l>>4)*8; elems 0..7 = K kb..kb+7, elems 8..15 = K kb+16..kb+23
__device__ __forceinline__ v16bf loadB16(const u16* __restrict__ base, int rowStride, int lane) {
  int n = lane & 15, kb = (lane >> 4) << 3;
  union { v16bf v; uint4 q[2]; } u;
  const u16* r = base + (size_t)n * rowStride + kb;
  u.q[0] = *(const uint4*)(r);
  u.q[1] = *(const uint4*)(r + 16);
  return u.v;
}

// ds_swizzle xor-butterfly within each 16-lane half (group-of-32 mode:
// offset[14:10]=xor_mask, [9:5]=or_mask, [4:0]=and_mask)
template <int XM>
__device__ __forceinline__ float swz_xor(float x) {
  return __int_as_float(
      __builtin_amdgcn_ds_swizzle(__float_as_int(x), (XM << 10) | 0x1f));
}

template <int XM>
__device__ __forceinline__ void rmax_step(float x[8]) {
#pragma unroll
  for (int r = 0; r < 8; ++r) x[r] = fmaxf(x[r], swz_xor<XM>(x[r]));
}
__device__ __forceinline__ void rmax8(float x[8]) {
  rmax_step<1>(x); rmax_step<2>(x); rmax_step<4>(x); rmax_step<8>(x);
}
template <int XM>
__device__ __forceinline__ void rsum_step(float x[8]) {
#pragma unroll
  for (int r = 0; r < 8; ++r) x[r] += swz_xor<XM>(x[r]);
}
__device__ __forceinline__ void rsum8(float x[8]) {
  rsum_step<1>(x); rsum_step<2>(x); rsum_step<4>(x); rsum_step<8>(x);
}

__device__ __forceinline__ float bias_ld(const float* __restrict__ w,
                                         const unsigned char* __restrict__ mk,
                                         int a, int b) {
  size_t i = (size_t)a * NB + b;
  if (!mk[i]) return -1e30f;
  return fix_io(w[i]);
}

// ---------------- kernel 1: QKV projection (fp32 math -> bf16 tiles) ----------------
__global__ void prep_qkv(const float* __restrict__ a_z, const float* __restrict__ bv_z,
                         const float* __restrict__ Wq, const float* __restrict__ Wk,
                         const float* __restrict__ Wv,
                         const float* __restrict__ bq, const float* __restrict__ bk,
                         const float* __restrict__ bvb,
                         u16* __restrict__ Qpad, u16* __restrict__ Kpad,
                         u16* __restrict__ Vt) {
  __shared__ float xr[DIM];
  int row = blockIdx.x, t = threadIdx.x;
  int mode, r;
  const float *x, *W, *bias;
  if (row < NA)            { mode = 0; r = row;            x = a_z + (size_t)r * DIM;  W = Wq; bias = bq; }
  else if (row < NA + NB)  { mode = 1; r = row - NA;       x = bv_z + (size_t)r * DIM; W = Wk; bias = bk; }
  else                     { mode = 2; r = row - NA - NB;  x = bv_z + (size_t)r * DIM; W = Wv; bias = bvb; }

  for (int i = t; i < DIM; i += 64) xr[i] = fix_io(x[i]);
  __syncthreads();

  const float sc = (mode == 0) ? rsqrtf((float)DK) : 1.f;
  for (int d = t; d < DIM; d += 64) {
    const float* wr = W + (size_t)d * DIM;
    float acc = 0.f;
#pragma unroll 8
    for (int e = 0; e < DIM; ++e) acc = fmaf(xr[e], wr[e], acc);
    acc += bias[d];
    if (mode == 0)      Qpad[(size_t)r * (NH * DKP) + (d / DK) * DKP + (d % DK)] = f2bf(acc * sc);
    else if (mode == 1) Kpad[(size_t)r * (NH * DKP) + (d / DK) * DKP + (d % DK)] = f2bf(acc);
    else                Vt[(size_t)d * NB + r] = f2bf(acc);
  }
  if (mode < 2) {  // zero the 4x16 pad slots (t = 0..63)
    u16* dst = ((mode == 0) ? Qpad : Kpad) + (size_t)r * (NH * DKP);
    dst[(t >> 4) * DKP + DK + (t & 15)] = 0;
  }
}

// ---------------- kernel 2: head-averaged Wo (folds mean over heads) ----------------
__global__ void prep_wo(const float* __restrict__ Wo, const float* __restrict__ bo,
                        u16* __restrict__ WoAvg, float* __restrict__ boAvg) {
  int gid = blockIdx.x * 256 + threadIdx.x;
  if (gid < DK * DIM) {
    int j = gid / DIM, e = gid % DIM;
    float s = 0.f;
#pragma unroll
    for (int h = 0; h < NH; ++h) s += Wo[(size_t)(h * DK + j) * DIM + e];
    WoAvg[(size_t)j * DIM + e] = f2bf(0.25f * s);
  }
  if (gid < DK) {
    float s = 0.f;
#pragma unroll
    for (int h = 0; h < NH; ++h) s += bo[h * DK + gid];
    boAvg[gid] = 0.25f * s;
  }
}

// prefetched-one-chunk-ahead operands: K B-tiles + additive bias only
// (V tiles are loaded just-in-time; they are consumed at the end of the body)
struct KB {
  v16bf k00, k01, k10, k11;   // K: subtile{0,1} x kchunk{0,1}
  float wb0[8], wb1[8];       // bias for the two 16-bill subtiles
};

// ---------------- kernel 3: flash attention over a bill segment ----------------
// grid (NA/16, SEG), block 32 (one wave). Heads outer; K/bias double-buffered.
__global__ void __launch_bounds__(32, 1)
attn_flash(const u16* __restrict__ Qpad, const u16* __restrict__ Kpad,
           const u16* __restrict__ Vt,
           const float* __restrict__ w, const unsigned char* __restrict__ mk,
           float* __restrict__ Part_o, float* __restrict__ Part_s) {
  __shared__ __align__(16) u16 pst[16 * CHUNK];   // staged probabilities (A operand)
  const int lane = threadIdx.x;
  const int atile = blockIdx.x * 16;
  const int seg = blockIdx.y;
  const int half = lane >> 4, col = lane & 15;
  const int abase = atile + half * 8;
  const int b0seg = seg * BSEG;

  float mw[8], Zw[8];
#pragma unroll
  for (int r = 0; r < 8; ++r) { mw[r] = -1e30f; Zw[r] = 0.f; }

  // one head pass; FIRST=true also maintains the w-softmax state (mw/Zw/betaw)
  auto head_pass = [&](auto FC, int h) {
    constexpr bool FIRST = decltype(FC)::value;

    const v16bf Aq0 = loadA16(Qpad + (size_t)atile * (NH * DKP) + h * DKP,       NH * DKP, lane);
    const v16bf Aq1 = loadA16(Qpad + (size_t)atile * (NH * DKP) + h * DKP + 32,  NH * DKP, lane);

    float m[8], L[8], T[8];
    v8f o0 = {}, o1 = {}, o2 = {};
#pragma unroll
    for (int r = 0; r < 8; ++r) { m[r] = -1e30f; L[r] = 0.f; T[r] = 0.f; }

    auto load_kb = [&](int b0) -> KB {
      KB ck;
      const u16* kb = Kpad + (size_t)b0 * (NH * DKP) + h * DKP;
      ck.k00 = loadB16(kb,                        NH * DKP, lane);
      ck.k01 = loadB16(kb + 32,                   NH * DKP, lane);
      ck.k10 = loadB16(kb + 16 * (NH * DKP),      NH * DKP, lane);
      ck.k11 = loadB16(kb + 16 * (NH * DKP) + 32, NH * DKP, lane);
#pragma unroll
      for (int r = 0; r < 8; ++r) {
        ck.wb0[r] = bias_ld(w, mk, abase + r, b0 + col);
        ck.wb1[r] = bias_ld(w, mk, abase + r, b0 + 16 + col);
      }
      return ck;
    };

    auto process_chunk = [&](const KB& ck, int b0) {
      // V tiles for THIS chunk: issue now, consume after the whole softmax section
      const u16* vb = Vt + (size_t)(h * DK) * NB + b0;
      const v16bf v0 = loadB16(vb,           NB, lane);
      const v16bf v1 = loadB16(vb + 16 * NB, NB, lane);
      const v16bf v2 = loadB16(vb + 32 * NB, NB, lane);

      float betaw[8];
      if constexpr (FIRST) {   // running softmax over w itself (norm_scores / influence)
        float wmx[8];
#pragma unroll
        for (int r = 0; r < 8; ++r) wmx[r] = fmaxf(ck.wb0[r], ck.wb1[r]);
        rmax8(wmx);
#pragma unroll
        for (int r = 0; r < 8; ++r) {
          float nm = fmaxf(mw[r], wmx[r]);
          betaw[r] = __expf(mw[r] - nm);
          mw[r] = nm;
        }
      }
      float ew0[8], ew1[8];
#pragma unroll
      for (int r = 0; r < 8; ++r) {
        ew0[r] = __expf(ck.wb0[r] - mw[r]);
        ew1[r] = __expf(ck.wb1[r] - mw[r]);
      }
      if constexpr (FIRST) {
        float za[8];
#pragma unroll
        for (int r = 0; r < 8; ++r) za[r] = ew0[r] + ew1[r];
        rsum8(za);
#pragma unroll
        for (int r = 0; r < 8; ++r) Zw[r] = Zw[r] * betaw[r] + za[r];
      }

      // ---- scores: q . k (K = 64 per head -> 2 WMMAs per 16x16 tile) ----
      v8f c0 = {}, c1 = {};
      c0 = __builtin_amdgcn_wmma_f32_16x16x32_bf16(false, Aq0, false, ck.k00,
                                                   (short)0, c0, false, false);
      c0 = __builtin_amdgcn_wmma_f32_16x16x32_bf16(false, Aq1, false, ck.k01,
                                                   (short)0, c0, false, false);
      c1 = __builtin_amdgcn_wmma_f32_16x16x32_bf16(false, Aq0, false, ck.k10,
                                                   (short)0, c1, false, false);
      c1 = __builtin_amdgcn_wmma_f32_16x16x32_bf16(false, Aq1, false, ck.k11,
                                                   (short)0, c1, false, false);

      float s0[8], s1[8];
#pragma unroll
      for (int r = 0; r < 8; ++r) { s0[r] = c0[r] + ck.wb0[r]; s1[r] = c1[r] + ck.wb1[r]; }

      // ---- flash update ----
      float mx[8];
#pragma unroll
      for (int r = 0; r < 8; ++r) mx[r] = fmaxf(s0[r], s1[r]);
      rmax8(mx);
      float al[8];
#pragma unroll
      for (int r = 0; r < 8; ++r) {
        float nm = fmaxf(m[r], mx[r]);
        al[r] = __expf(m[r] - nm);
        m[r] = nm;
      }
      float p0[8], p1[8];
#pragma unroll
      for (int r = 0; r < 8; ++r) {
        p0[r] = __expf(s0[r] - m[r]);
        p1[r] = __expf(s1[r] - m[r]);
      }
      float la[8];
#pragma unroll
      for (int r = 0; r < 8; ++r) la[r] = p0[r] + p1[r];
      rsum8(la);
#pragma unroll
      for (int r = 0; r < 8; ++r) L[r] = L[r] * al[r] + la[r];
      float ta[8];
#pragma unroll
      for (int r = 0; r < 8; ++r) ta[r] = p0[r] * ew0[r] + p1[r] * ew1[r];
      rsum8(ta);
      if constexpr (FIRST) {
#pragma unroll
        for (int r = 0; r < 8; ++r) T[r] = T[r] * al[r] * betaw[r] + ta[r];
      } else {
#pragma unroll
        for (int r = 0; r < 8; ++r) T[r] = T[r] * al[r] + ta[r];
      }

      // ---- stage p (bf16) to LDS in row-major [actor][bill] for the A operand ----
#pragma unroll
      for (int r = 0; r < 8; ++r) {
        pst[(r + half * 8) * CHUNK + col]      = f2bf(p0[r]);
        pst[(r + half * 8) * CHUNK + 16 + col] = f2bf(p1[r]);
      }
      const v16bf Ap = loadA16(pst, CHUNK, lane);

      v8f alv;
#pragma unroll
      for (int r = 0; r < 8; ++r) alv[r] = al[r];

      // ---- out += P * V  (3 d-tiles of 16, K = 32 bills) ----
      o0 = __builtin_amdgcn_wmma_f32_16x16x32_bf16(false, Ap, false, v0,
                                                   (short)0, o0 * alv, false, false);
      o1 = __builtin_amdgcn_wmma_f32_16x16x32_bf16(false, Ap, false, v1,
                                                   (short)0, o1 * alv, false, false);
      o2 = __builtin_amdgcn_wmma_f32_16x16x32_bf16(false, Ap, false, v2,
                                                   (short)0, o2 * alv, false, false);
    };

    // ---- ping-pong chunk pipeline (K/bias one chunk ahead, no rotation copies) ----
    KB ca = load_kb(b0seg);
#pragma unroll 1
    for (int c = 0; c < NCH; c += 2) {
      KB cb = load_kb(b0seg + (c + 1) * CHUNK);   // in flight during process(ca)
      process_chunk(ca, b0seg + c * CHUNK);
      if (c + 2 < NCH) ca = load_kb(b0seg + (c + 2) * CHUNK);  // in flight during process(cb)
      process_chunk(cb, b0seg + (c + 1) * CHUNK);
    }

    // ---- segment partials ----
#pragma unroll
    for (int r = 0; r < 8; ++r) {
      float* po = Part_o + ((size_t)(abase + r) * SEG + seg) * DIM + h * DK;
      po[col] = o0[r]; po[16 + col] = o1[r]; po[32 + col] = o2[r];
    }
    if (col == 0) {   // lanes 0 and 16 hold state for actors abase..abase+7
#pragma unroll
      for (int r = 0; r < 8; ++r) {
        float* ps = Part_s + ((size_t)(abase + r) * SEG + seg) * 16;
        ps[h] = m[r]; ps[4 + h] = L[r]; ps[8 + h] = T[r];
        if constexpr (FIRST) { ps[12] = mw[r]; ps[13] = Zw[r]; }
      }
    }
  };

  head_pass(BoolC<true>{}, 0);
#pragma unroll 1
  for (int h = 1; h < NH; ++h) head_pass(BoolC<false>{}, h);
}

// ---------------- kernel 4: merge segments + final WMMA projection ----------------
__global__ void __launch_bounds__(32, 1)
merge_out(const float* __restrict__ Part_o, const float* __restrict__ Part_s,
          const u16* __restrict__ WoAvg, const float* __restrict__ boAvg,
          float* __restrict__ out) {
  __shared__ __align__(16) u16 ost[16 * DIM];  // normalized attn out (bf16)
  __shared__ float scl[16][SEG][NH];
  __shared__ float invL[16][NH];
  const int lane = threadIdx.x;
  const int atile = blockIdx.x * 16;

  if (lane < 16) {
    const int a = atile + lane;
    const float* ps = Part_s + (size_t)a * SEG * 16;
    float mwS = -1e30f;
    for (int s = 0; s < SEG; ++s) mwS = fmaxf(mwS, ps[s * 16 + 12]);
    float ZwS = 0.f;
    for (int s = 0; s < SEG; ++s) ZwS += ps[s * 16 + 13] * __expf(ps[s * 16 + 12] - mwS);
    float infl = 0.f;
    for (int h = 0; h < NH; ++h) {
      float mS = -1e30f;
      for (int s = 0; s < SEG; ++s) mS = fmaxf(mS, ps[s * 16 + h]);
      float LS = 0.f, TS = 0.f;
      for (int s = 0; s < SEG; ++s) {
        float e = __expf(ps[s * 16 + h] - mS);
        scl[lane][s][h] = e;
        LS += ps[s * 16 + 4 + h] * e;
        TS += ps[s * 16 + 8 + h] * e * __expf(ps[s * 16 + 12] - mwS);
      }
      float iL = 1.f / LS;
      invL[lane][h] = iL;
      infl += TS * iL;
    }
    infl = 0.25f * infl / ZwS;
    if (infl != infl) infl = 0.f;
    else if (infl == __builtin_inff()) infl = 1.f;
    else if (infl == -__builtin_inff()) infl = -1.f;
    out[(size_t)a * (DK + 1) + DK] = infl;
  }
  __syncthreads();

  for (int idx = lane; idx < 16 * DIM; idx += 32) {
    int a16 = idx / DIM, d = idx % DIM, h = d / DK;
    const float* po = Part_o + (size_t)(atile + a16) * SEG * DIM + d;
    float acc = 0.f;
    for (int s = 0; s < SEG; ++s) acc += po[(size_t)s * DIM] * scl[a16][s][h];
    ost[a16 * DIM + d] = f2bf(acc * invL[a16][h]);
  }
  __syncthreads();

  // topic_align = out_norm @ WoAvg^T + boAvg  (16x48, K=192 -> 3 j-tiles x 6 WMMAs)
  for (int jt = 0; jt < 3; ++jt) {
    v8f c = {};
#pragma unroll
    for (int kc = 0; kc < 6; ++kc) {
      c = __builtin_amdgcn_wmma_f32_16x16x32_bf16(false,
            loadA16(ost + kc * 32, DIM, lane), false,
            loadB16(WoAvg + (size_t)jt * 16 * DIM + kc * 32, DIM, lane),
            (short)0, c, false, false);
    }
    const int j = jt * 16 + (lane & 15);
    const float bo = boAvg[j];
#pragma unroll
    for (int r = 0; r < 8; ++r) {
      const int a = atile + r + (lane >> 4) * 8;
      float v = c[r] + bo;
      if (!(v == v) || v == __builtin_inff() || v == -__builtin_inff()) v = 0.f;
      out[(size_t)a * (DK + 1) + j] = v;
    }
  }
}

// ---------------- launch ----------------
extern "C" void kernel_launch(void* const* d_in, const int* in_sizes, int n_in,
                              void* d_out, int out_size, void* d_ws, size_t ws_size,
                              hipStream_t stream) {
  const float* a_z  = (const float*)d_in[0];
  const float* bv_z = (const float*)d_in[1];
  const float* w    = (const float*)d_in[2];
  const unsigned char* mk = (const unsigned char*)d_in[3];  // jax bool -> 1 byte
  const float* Wq = (const float*)d_in[4];
  const float* Wk = (const float*)d_in[5];
  const float* Wv = (const float*)d_in[6];
  const float* Wo = (const float*)d_in[7];
  const float* bq = (const float*)d_in[8];
  const float* bk = (const float*)d_in[9];
  const float* bvb = (const float*)d_in[10];
  const float* bo = (const float*)d_in[11];
  float* out = (float*)d_out;

  char* ws = (char*)d_ws;
  size_t off = 0;
  auto take = [&](size_t bytes) {
    char* p = ws + off;
    off = (off + bytes + 255) & ~(size_t)255;
    return p;
  };
  u16*   Qpad   = (u16*)take((size_t)NA * NH * DKP * sizeof(u16));
  u16*   Kpad   = (u16*)take((size_t)NB * NH * DKP * sizeof(u16));
  u16*   Vt     = (u16*)take((size_t)DIM * NB * sizeof(u16));
  u16*   WoAvg  = (u16*)take((size_t)DK * DIM * sizeof(u16));
  float* boAvg  = (float*)take((size_t)DK * sizeof(float));
  float* Part_o = (float*)take((size_t)NA * SEG * DIM * sizeof(float));
  float* Part_s = (float*)take((size_t)NA * SEG * 16 * sizeof(float));
  (void)ws_size; (void)n_in; (void)in_sizes; (void)out_size;

  prep_qkv<<<NA + 2 * NB, 64, 0, stream>>>(a_z, bv_z, Wq, Wk, Wv, bq, bk, bvb,
                                           Qpad, Kpad, Vt);
  prep_wo<<<(DK * DIM + 255) / 256, 256, 0, stream>>>(Wo, bo, WoAvg, boAvg);
  attn_flash<<<dim3(NA / 16, SEG), 32, 0, stream>>>(Qpad, Kpad, Vt, w, mk,
                                                    Part_o, Part_s);
  merge_out<<<NA / 16, 32, 0, stream>>>(Part_o, Part_s, WoAvg, boAvg, out);
}
